// Attention_loss_919123001759
// MI455X (gfx1250) — matrix operand order repaired and checked by reading, not compiled, not run on hardware
//
#include <hip/hip_runtime.h>
#include <hip/hip_bf16.h>

typedef __attribute__((ext_vector_type(16))) _Float16 v16h;
typedef __attribute__((ext_vector_type(8)))  _Float16 v8h;
typedef __attribute__((ext_vector_type(4)))  _Float16 v4h;
typedef __attribute__((ext_vector_type(8)))  float    v8f;

#define NN   8192      // 2*B
#define BH   4096      // B
#define DD   1024      // D
#define BM   128
#define BN   128
#define BK   32
#define LDSP 40        // padded LDS row stride in halves (80 B, keeps 16B alignment)

// ---------------- kernel 1: f32 -> f16, concat(f1,f2) ----------------
__global__ void convert_kernel(const float* __restrict__ f1,
                               const float* __restrict__ f2,
                               _Float16* __restrict__ H) {
  int idx = blockIdx.x * blockDim.x + threadIdx.x;   // float4 index
  const int halfTot = BH * DD / 4;
  const float4* src = (idx < halfTot) ? ((const float4*)f1 + idx)
                                      : ((const float4*)f2 + (idx - halfTot));
  float4 v = *src;
  v4h h;
  h.x = (_Float16)v.x; h.y = (_Float16)v.y;
  h.z = (_Float16)v.z; h.w = (_Float16)v.w;
  *((v4h*)H + idx) = h;
}

// ---------------- kernel 2: zero the num/den accumulators ----------------
__global__ void zero_kernel(float* __restrict__ p, int n) {
  int idx = blockIdx.x * blockDim.x + threadIdx.x;
  if (idx < n) p[idx] = 0.0f;
}

// async 16B copy: global -> LDS, tracked by ASYNCcnt (CDNA5 path, no VGPR staging)
__device__ __forceinline__ void async_cp16(unsigned lds_addr, const _Float16* gsrc) {
  asm volatile("global_load_async_to_lds_b128 %0, %1, off"
               :: "v"(lds_addr), "v"(gsrc) : "memory");
}

__device__ __forceinline__ v16h frag16(const _Float16* p, int lo_off, int hi_off) {
  v8h lo = *(const v8h*)(p + lo_off);
  v8h hi = *(const v8h*)(p + hi_off);
  return __builtin_shufflevector(lo, hi, 0,1,2,3,4,5,6,7,8,9,10,11,12,13,14,15);
}

// ---------------- kernel 3: fused WMMA H*H^T + exp + masked row sums ----
__global__ __launch_bounds__(256)
void sim_loss_kernel(const _Float16* __restrict__ H,   // [NN, DD] f16
                     const float*   __restrict__ Sw,   // [BH, BH] f32
                     const int*     __restrict__ label,// [BH]
                     float* __restrict__ num,          // [NN]
                     float* __restrict__ den) {        // [NN]
  __shared__ __align__(16) _Float16 As[2][BM * LDSP];
  __shared__ __align__(16) _Float16 Bs[2][BN * LDSP];

  const int i0   = blockIdx.y * BM;
  const int j0   = blockIdx.x * BN;
  const int tid  = threadIdx.x;
  const int wave = tid >> 5;
  const int wm   = wave & 3;       // 4 waves along M: rows [wm*32, wm*32+32)
  const int wn   = wave >> 2;      // 2 waves along N: cols [wn*64, wn*64+64)
  const int lane = tid & 31;
  const int n    = lane & 15;      // row/col within a 16x16 tile
  const int hw   = lane >> 4;      // half-wave select

  // cooperative copy pattern: thread owns rows r0 and r0+64, 16B column chunk c0
  const int r0 = tid >> 2;
  const int c0 = (tid & 3) * 8;

  const unsigned lA[2] = { (unsigned)(uintptr_t)&As[0][r0 * LDSP + c0],
                           (unsigned)(uintptr_t)&As[1][r0 * LDSP + c0] };
  const unsigned lB[2] = { (unsigned)(uintptr_t)&Bs[0][r0 * LDSP + c0],
                           (unsigned)(uintptr_t)&Bs[1][r0 * LDSP + c0] };
  const unsigned rowHop = 64u * LDSP * sizeof(_Float16);

  auto issue = [&](int buf, int k0) {
    const _Float16* gA0 = &H[(size_t)(i0 + r0) * DD + k0 + c0];
    const _Float16* gB0 = &H[(size_t)(j0 + r0) * DD + k0 + c0];
    async_cp16(lA[buf],          gA0);
    async_cp16(lA[buf] + rowHop, gA0 + (size_t)64 * DD);
    async_cp16(lB[buf],          gB0);
    async_cp16(lB[buf] + rowHop, gB0 + (size_t)64 * DD);
  };

  v8f acc[2][4];
  #pragma unroll
  for (int mt = 0; mt < 2; ++mt)
    #pragma unroll
    for (int nt = 0; nt < 4; ++nt) acc[mt][nt] = (v8f)0.0f;

  issue(0, 0);                                 // prologue: tile 0 in flight
  int buf = 0;
  for (int k0 = 0; k0 < DD; k0 += BK) {
    asm volatile("s_wait_asynccnt 0x0" ::: "memory");  // my copies for this tile landed
    __syncthreads();                                    // everyone's landed; other buffer free
    if (k0 + BK < DD) issue(buf ^ 1, k0 + BK);          // overlap next copy with compute

    // A fragments (ISA 7.12.2, 16-bit A 16x32): lane row m=(lane&15),
    // k = [hw*8, hw*8+8) ∪ [16+hw*8, 16+hw*8+8)
    v16h a[2];
    #pragma unroll
    for (int mt = 0; mt < 2; ++mt)
      a[mt] = frag16(&As[buf][(wm * 32 + mt * 16 + n) * LDSP], hw * 8, 16 + hw * 8);

    #pragma unroll
    for (int nt = 0; nt < 4; ++nt) {
      // B fragment (j-panel transposed): lane col n=(lane&15), k = [hw*8, hw*8+8) ∪ [16+hw*8, 16+hw*8+8)
      v16h b = frag16(&Bs[buf][(wn * 64 + nt * 16 + n) * LDSP], hw * 8, 16 + hw * 8);
      acc[0][nt] = __builtin_amdgcn_wmma_f32_16x16x32_f16(
          false, a[0], false, b, (short)0, acc[0][nt], false, false);
      acc[1][nt] = __builtin_amdgcn_wmma_f32_16x16x32_f16(
          false, a[1], false, b, (short)0, acc[1][nt], false, false);
    }
    buf ^= 1;
  }

  // ---- fused epilogue: exp, masks, row reduction ----
  int labj[4];
  #pragma unroll
  for (int nt = 0; nt < 4; ++nt)
    labj[nt] = label[(j0 + wn * 64 + nt * 16 + n) & (BH - 1)];

  #pragma unroll
  for (int mt = 0; mt < 2; ++mt) {
    #pragma unroll
    for (int r = 0; r < 8; ++r) {
      const int i    = i0 + wm * 32 + mt * 16 + r + 8 * hw; // C/D: VGPR r -> M=r or 8+r
      const int li   = i & (BH - 1);
      const int labi = label[li];
      float numacc = 0.0f, denacc = 0.0f;
      #pragma unroll
      for (int nt = 0; nt < 4; ++nt) {
        const int j = j0 + wn * 64 + nt * 16 + n;
        const float e = __expf(acc[mt][nt][r] * 20.0f);   // exp(sim / 0.05)
        int d = i - j; d = (d < 0) ? -d : d;
        if (d != 0 && d != BH) denacc += e;               // mask_correlated
        if (labi == labj[nt]) {                            // category match (~1%)
          const int lj = j & (BH - 1);
          float s = (i == j) ? 0.0f : Sw[(size_t)li * BH + lj];
          numacc += e * (s + 1e-5f);
        }
      }
      // reduce across the 16 lanes of this half-wave (bits 0..3 of lane id)
      #pragma unroll
      for (int off = 1; off < 16; off <<= 1) {
        numacc += __shfl_xor(numacc, off, 32);
        denacc += __shfl_xor(denacc, off, 32);
      }
      if (n == 0) {
        atomicAdd(&num[i], numacc);
        atomicAdd(&den[i], denacc);
      }
    }
  }
}

// ---------------- kernel 4: loss = -mean(log(num/den)) ----------------
__global__ void finalize_kernel(const float* __restrict__ num,
                                const float* __restrict__ den,
                                float* __restrict__ out) {
  __shared__ float red[256];
  float acc = 0.0f;
  for (int i = threadIdx.x; i < NN; i += 256)
    acc += __logf(num[i] / den[i]);
  red[threadIdx.x] = acc;
  __syncthreads();
  for (int s = 128; s > 0; s >>= 1) {
    if (threadIdx.x < s) red[threadIdx.x] += red[threadIdx.x + s];
    __syncthreads();
  }
  if (threadIdx.x == 0) out[0] = -red[0] / (float)NN;
}

extern "C" void kernel_launch(void* const* d_in, const int* in_sizes, int n_in,
                              void* d_out, int out_size, void* d_ws, size_t ws_size,
                              hipStream_t stream) {
  const float* f1    = (const float*)d_in[0];
  const float* f2    = (const float*)d_in[1];
  const float* Sw    = (const float*)d_in[2];
  const int*   label = (const int*)d_in[3];
  float* out = (float*)d_out;

  _Float16* H  = (_Float16*)d_ws;
  float* num   = (float*)((char*)d_ws + (size_t)NN * DD * sizeof(_Float16));
  float* den   = num + NN;

  convert_kernel<<<NN * DD / 4 / 256, 256, 0, stream>>>(f1, f2, H);
  zero_kernel<<<(2 * NN + 255) / 256, 256, 0, stream>>>(num, 2 * NN);
  dim3 grid(NN / BN, NN / BM);
  sim_loss_kernel<<<grid, 256, 0, stream>>>(H, Sw, label, num, den);
  finalize_kernel<<<1, 256, 0, stream>>>(num, den, out);
}
